// TKANCell_44719199486012
// MI455X (gfx1250) — matrix operand order repaired and checked by reading, not compiled, not run on hardware
//
#include <hip/hip_runtime.h>
#include <cstdint>
#include <cstddef>

// ---------- types ----------
typedef __attribute__((ext_vector_type(16))) __bf16 v16bf;
typedef __attribute__((ext_vector_type(8)))  __bf16 v8bf;
typedef __attribute__((ext_vector_type(8)))  float  v8f;

// ---------- problem sizes ----------
#define B_SZ   2048
#define D_SZ   1024
#define U_SZ   1024
#define L_SZ   3
#define NB     8          // NUM_BASIS = GRID_SIZE + SPLINE_ORDER = 8

// ---------- GEMM tiling ----------
constexpr int BM = 128, BN = 128, BK = 32, TPB = 256;
constexpr int LDP = 40;   // padded LDS row stride (80B) -> bank-conflict-free b128 frag loads

__device__ __forceinline__ unsigned pack2_bf16(float lo, float hi) {
  unsigned short a = __builtin_bit_cast(unsigned short, (__bf16)lo);
  unsigned short b = __builtin_bit_cast(unsigned short, (__bf16)hi);
  return (unsigned)a | ((unsigned)b << 16);
}

__device__ __forceinline__ float sigmoidf_(float v) {
  return 1.0f / (1.0f + __expf(-v));
}

// 16 consecutive elements -> two v8bf chunks (bf16 passthrough / f32 convert)
__device__ __forceinline__ void load16_bf16(const __bf16* p, v8bf& c0, v8bf& c1) {
  c0 = *(const v8bf*)p;
  c1 = *(const v8bf*)(p + 8);
}
__device__ __forceinline__ void load16_bf16(const float* p, v8bf& c0, v8bf& c1) {
#pragma unroll
  for (int i = 0; i < 8; ++i) { c0[i] = (__bf16)p[i]; c1[i] = (__bf16)p[i + 8]; }
}

// =====================================================================
// prep: bf16 casts of x / h_prev / silu(x), and cubic B-spline basis
// (Cox-de Boor, uniform knots t_i = -1 + 0.4*(i-3), i = 0..11)
// =====================================================================
__global__ __launch_bounds__(256)
void prep_kernel(const float* __restrict__ x, const float* __restrict__ h_prev,
                 __bf16* __restrict__ xb, __bf16* __restrict__ hb,
                 __bf16* __restrict__ silub, __bf16* __restrict__ basesb,
                 int total)
{
  int idx = blockIdx.x * blockDim.x + threadIdx.x;
  if (idx >= total) return;

  float xv = x[idx];
  xb[idx]    = (__bf16)xv;
  hb[idx]    = (__bf16)h_prev[idx];
  silub[idx] = (__bf16)(xv * sigmoidf_(xv));

  float bb[11];
#pragma unroll
  for (int i = 0; i < 11; ++i) {
    float ti  = -1.0f + 0.4f * (float)(i - 3);
    bb[i] = (xv >= ti && xv < ti + 0.4f) ? 1.0f : 0.0f;
  }
#pragma unroll
  for (int p = 1; p <= 3; ++p) {
    float inv = 1.0f / (0.4f * (float)p);
#pragma unroll
    for (int i = 0; i < 10; ++i) {
      if (i < 11 - p) {
        float ti  = -1.0f + 0.4f * (float)(i - 3);
        float tip = ti + 0.4f * (float)p;
        bb[i] = (xv - ti) * inv * bb[i] + (tip + 0.4f - xv) * inv * bb[i + 1];
      }
    }
  }
  size_t base = (size_t)idx * NB;
#pragma unroll
  for (int k = 0; k < NB; ++k) basesb[base + k] = (__bf16)bb[k];
}

// =====================================================================
// LSTM gate nonlinearity: z -> (h, c), also scatter h into combined[:,0:U]
// =====================================================================
__global__ __launch_bounds__(256)
void gates_kernel(const float* __restrict__ z, const float* __restrict__ c_prev,
                  float* __restrict__ h_out, float* __restrict__ c_out,
                  float* __restrict__ combined)
{
  int idx = blockIdx.x * blockDim.x + threadIdx.x;
  if (idx >= B_SZ * U_SZ) return;
  int b = idx / U_SZ, u = idx - b * U_SZ;
  const float* zr = z + (size_t)b * (4 * U_SZ);
  float zi = zr[u];
  float zf = zr[U_SZ + u];
  float zg = zr[2 * U_SZ + u];
  float zo = zr[3 * U_SZ + u];
  float c = sigmoidf_(zf) * c_prev[idx] + sigmoidf_(zi) * tanhf(zg);
  float h = sigmoidf_(zo) * tanhf(c);
  c_out[idx] = c;
  h_out[idx] = h;
  combined[(size_t)b * (4 * U_SZ) + u] = h;
}

// =====================================================================
// bf16 WMMA GEMM:  C[M,N] = A1[M,K1]*W1[K1,N] (+ A2[M,K2]*W2[K2,N]) (+ bias)
//  * A tile staged in LDS in natural K order [0..31] so each
//    lane's A-fragment is two contiguous ds_load_b128
//  * W tile staged transposed BT[n][k] so each lane's B-fragment is two
//    contiguous ds_load_b128 (B order used here: lane=col, VGPR j = K khalf*16+2j,2j+1)
//  * software pipeline: next HBM tile fetched into regs during compute
//  * blockIdx.z selects layer via w/c z-strides (fused KAN layers)
// =====================================================================
template <typename AT>
__device__ __forceinline__ void fetch_tiles(
    int it, int nt1,
    const AT* __restrict__ A1, int lda1, const float* __restrict__ W1, int ldw1,
    const AT* __restrict__ A2, int lda2, const float* __restrict__ W2, int ldw2,
    int m0, int n0, int tid,
    v8bf& a0, v8bf& a1, unsigned (&bp)[8])
{
  const bool second = it >= nt1;
  const AT*    A   = second ? A2 : A1;
  const float* W   = second ? W2 : W1;
  const int    lda = second ? lda2 : lda1;
  const int    ldw = second ? ldw2 : ldw1;
  const int    k0  = (second ? it - nt1 : it) * BK;
  // A: 2 threads per row, 16 consecutive K each
  {
    const int r  = tid >> 1;
    const int cg = (tid & 1) * 16;
    load16_bf16(A + (size_t)(m0 + r) * lda + k0 + cg, a0, a1);
  }
  // W: K-pair rows (2j, 2j+1), 8 columns each, packed bf16 pairs
  {
    const int j   = tid & 15;
    const int ncg = (tid >> 4) * 8;
    const float* s0 = W + (size_t)(k0 + 2 * j) * ldw + n0 + ncg;
    const float* s1 = s0 + ldw;
#pragma unroll
    for (int i = 0; i < 8; ++i) bp[i] = pack2_bf16(s0[i], s1[i]);
  }
}

template <typename AT>
__global__ __launch_bounds__(TPB)
void gemm_bf16_wmma(const AT* __restrict__ A1, int lda1, int K1,
                    const float* __restrict__ W1, int ldw1, size_t w1_zs,
                    const AT* __restrict__ A2, int lda2, int K2,
                    const float* __restrict__ W2, int ldw2, size_t w2_zs,
                    float* __restrict__ C, int ldc, size_t c_zs,
                    const float* __restrict__ bias)
{
  __shared__ __bf16 As[BM][LDP];   // 10 KB, [m][k]
  __shared__ __bf16 BT[BN][LDP];   // 10 KB, [n][k]

  W1 += (size_t)blockIdx.z * w1_zs;
  if (W2) W2 += (size_t)blockIdx.z * w2_zs;
  C  += (size_t)blockIdx.z * c_zs;

  const int tid  = threadIdx.x;
  const int lane = tid & 31;
  const int w    = tid >> 5;
  const int wm   = w & 3;        // wave M group (4 x 32 rows)
  const int wn   = w >> 2;       // wave N group (2 x 64 cols)
  const int m0   = blockIdx.y * BM;
  const int n0   = blockIdx.x * BN;
  const int khalf = lane >> 4;
  const int l16   = lane & 15;
  const int khoff = khalf * 16;

  v8f acc[2][4];
#pragma unroll
  for (int i = 0; i < 2; ++i)
#pragma unroll
    for (int j = 0; j < 4; ++j)
#pragma unroll
      for (int e = 0; e < 8; ++e) acc[i][j][e] = 0.0f;

  const int nt1 = K1 / BK;
  const int nt2 = A2 ? (K2 / BK) : 0;
  const int NT  = nt1 + nt2;

  // LDS write coordinates for this thread
  const int wr  = tid >> 1;
  const int wp0 = (tid & 1) ? 16 : 0;      // contiguous chunk positions: p0 and p0+8
  const int wj  = tid & 15;
  const int wncg = (tid >> 4) * 8;

  v8bf a_st0, a_st1;
  unsigned b_st[8];
  fetch_tiles<AT>(0, nt1, A1, lda1, W1, ldw1, A2, lda2, W2, ldw2, m0, n0, tid,
                  a_st0, a_st1, b_st);

  for (int it = 0; it < NT; ++it) {
    __syncthreads();   // all reads of previous tile complete
    // stage registers -> LDS
    *(v8bf*)&As[wr][wp0]      = a_st0;
    *(v8bf*)&As[wr][wp0 + 8]  = a_st1;
#pragma unroll
    for (int i = 0; i < 8; ++i)
      *(unsigned*)&BT[wncg + i][2 * wj] = b_st[i];
    __syncthreads();

    // prefetch next tile from HBM while computing this one
    if (it + 1 < NT)
      fetch_tiles<AT>(it + 1, nt1, A1, lda1, W1, ldw1, A2, lda2, W2, ldw2,
                      m0, n0, tid, a_st0, a_st1, b_st);

#pragma unroll
    for (int fm = 0; fm < 2; ++fm) {
      const __bf16* ap = &As[wm * 32 + fm * 16 + l16][khoff];
      v8bf ax0 = *(const v8bf*)ap;
      v8bf ax1 = *(const v8bf*)(ap + 8);
      v16bf afrag = __builtin_shufflevector(ax0, ax1,
          0, 1, 2, 3, 4, 5, 6, 7, 8, 9, 10, 11, 12, 13, 14, 15);
#pragma unroll
      for (int fn = 0; fn < 4; ++fn) {
        const __bf16* bpp = &BT[wn * 64 + fn * 16 + l16][khoff];
        v8bf bx0 = *(const v8bf*)bpp;
        v8bf bx1 = *(const v8bf*)(bpp + 8);
        v16bf bfrag = __builtin_shufflevector(bx0, bx1,
            0, 1, 2, 3, 4, 5, 6, 7, 8, 9, 10, 11, 12, 13, 14, 15);
        acc[fm][fn] = __builtin_amdgcn_wmma_f32_16x16x32_bf16(
            false, afrag, false, bfrag, (short)0, acc[fm][fn], false, false);
      }
    }
  }

  // ---- store C (+bias): lane l16 = N col; VGPR v = row v + 8*khalf ----
#pragma unroll
  for (int fm = 0; fm < 2; ++fm)
#pragma unroll
    for (int fn = 0; fn < 4; ++fn) {
      const int col = n0 + wn * 64 + fn * 16 + l16;
      const float bv = bias ? bias[col] : 0.0f;
#pragma unroll
      for (int v = 0; v < 8; ++v) {
        const int row = m0 + wm * 32 + fm * 16 + 8 * khalf + v;
        C[(size_t)row * ldc + col] = acc[fm][fn][v] + bv;
      }
    }
}

// =====================================================================
// launch
// =====================================================================
extern "C" void kernel_launch(void* const* d_in, const int* in_sizes, int n_in,
                              void* d_out, int out_size, void* d_ws, size_t ws_size,
                              hipStream_t stream) {
  const float* x        = (const float*)d_in[0];
  const float* h_prev   = (const float*)d_in[1];
  const float* c_prev   = (const float*)d_in[2];
  const float* lstm_k   = (const float*)d_in[3];
  const float* lstm_r   = (const float*)d_in[4];
  const float* lstm_b   = (const float*)d_in[5];
  const float* kan_bw   = (const float*)d_in[6];
  const float* kan_sw   = (const float*)d_in[7];
  const float* comb_w   = (const float*)d_in[8];
  const float* comb_b   = (const float*)d_in[9];

  float* out_y = (float*)d_out;                                // [B, U]
  float* out_h = out_y + (size_t)B_SZ * U_SZ;                  // [B, U]
  float* out_c = out_h + (size_t)B_SZ * U_SZ;                  // [B, U]

  // workspace carve-up
  char* ws = (char*)d_ws;
  float*  z        = (float*)ws;      ws += (size_t)B_SZ * 4 * U_SZ * 4;   // 32 MB
  float*  combined = (float*)ws;      ws += (size_t)B_SZ * 4 * U_SZ * 4;   // 32 MB
  __bf16* xb       = (__bf16*)ws;     ws += (size_t)B_SZ * D_SZ * 2;       // 4 MB
  __bf16* hb       = (__bf16*)ws;     ws += (size_t)B_SZ * U_SZ * 2;       // 4 MB
  __bf16* silub    = (__bf16*)ws;     ws += (size_t)B_SZ * D_SZ * 2;       // 4 MB
  __bf16* basesb   = (__bf16*)ws;     /* [B, D*8] bf16, 32 MB */

  // 1) elementwise prep
  {
    int total = B_SZ * D_SZ;
    prep_kernel<<<(total + 255) / 256, 256, 0, stream>>>(x, h_prev, xb, hb, silub, basesb, total);
  }

  // 2) z = x*Wk + h_prev*Wr + bias   [2048 x 4096]
  {
    dim3 grid(4 * U_SZ / BN, B_SZ / BM, 1);
    gemm_bf16_wmma<__bf16><<<grid, TPB, 0, stream>>>(
        xb, D_SZ, D_SZ, lstm_k, 4 * U_SZ, 0,
        hb, U_SZ, U_SZ, lstm_r, 4 * U_SZ, 0,
        z, 4 * U_SZ, 0, lstm_b);
  }

  // 3) gates -> h, c, combined[:,0:U]
  {
    int total = B_SZ * U_SZ;
    gates_kernel<<<(total + 255) / 256, 256, 0, stream>>>(z, c_prev, out_h, out_c, combined);
  }

  // 4) KAN layers (fused over blockIdx.z = l):
  //    combined[:, (1+l)*U] = silu(x)*Wb[l] + bases*Ws[l]
  {
    dim3 grid(U_SZ / BN, B_SZ / BM, L_SZ);
    gemm_bf16_wmma<__bf16><<<grid, TPB, 0, stream>>>(
        silub,  D_SZ,      D_SZ,      kan_bw, U_SZ, (size_t)D_SZ * U_SZ,
        basesb, D_SZ * NB, D_SZ * NB, kan_sw, U_SZ, (size_t)D_SZ * NB * U_SZ,
        combined + U_SZ, 4 * U_SZ, (size_t)U_SZ, nullptr);
  }

  // 5) output = combined * combine_w + combine_b
  {
    dim3 grid(U_SZ / BN, B_SZ / BM, 1);
    gemm_bf16_wmma<float><<<grid, TPB, 0, stream>>>(
        combined, 4 * U_SZ, 4 * U_SZ, comb_w, U_SZ, 0,
        (const float*)nullptr, 0, 0, nullptr, 0, 0,
        out_y, U_SZ, 0, comb_b);
  }
}
